// IntrinsicReward_3393024164556
// MI455X (gfx1250) — matrix operand order repaired and checked by reading, not compiled, not run on hardware
//
#include <hip/hip_runtime.h>

// ---------- problem constants (match reference) ----------
#define BQ   1024
#define DD   256
#define AAC  6
#define MM   200000
#define TOPK 16
#define H1C  128
#define H2C  64
#define LN_EPS 1e-5f

// GEMM tiling: 8 B-blocks of 128 rows; M covered by SPLITS splits of
// CHUNKS_PER_SPLIT chunks of 128 mem rows. 121*13*128 = 201344 >= 200000.
#define SPLITS 121
#define CHUNKS_PER_SPLIT 13
#define TILE_BYTES (128 * 256 * 2)   // one mem tile: 128 rows x 256 bf16 = 64KB

typedef __attribute__((ext_vector_type(16))) __bf16 v16bf;
typedef __attribute__((ext_vector_type(8)))  float  v8f;
typedef unsigned int u32x4 __attribute__((ext_vector_type(4)));
typedef int          i32x8 __attribute__((ext_vector_type(8)));
typedef int          i32x4 __attribute__((ext_vector_type(4)));

struct alignas(16) Q16 { unsigned int x, y, z, w; };
union Frag { v16bf v; Q16 q[2]; };

#if defined(__has_builtin)
#if __has_builtin(__builtin_amdgcn_tensor_load_to_lds) && __has_builtin(__builtin_amdgcn_s_wait_tensorcnt)
#define USE_TDM 1
#endif
#endif

// ---------- row-normalize + convert to bf16 (folds 1/(||x||+1e-8)) ----------
__global__ void prep_norm_bf16(const float* __restrict__ src,
                               __bf16* __restrict__ dst) {
  const int row = blockIdx.x;
  const int tid = threadIdx.x;              // 256 threads == DD
  __shared__ float red[256];
  float v = src[(size_t)row * DD + tid];
  red[tid] = v * v;
  __syncthreads();
  for (int s = 128; s > 0; s >>= 1) {
    if (tid < s) red[tid] += red[tid + s];
    __syncthreads();
  }
  const float invn = 1.0f / (sqrtf(red[0]) + 1e-8f);
  dst[(size_t)row * DD + tid] = (__bf16)(v * invn);
}

// ---------- forward model -> pred_err (out row 1) ----------
__global__ void forward_pred(const float* __restrict__ z_t,
                             const float* __restrict__ action,
                             const float* __restrict__ z_t1,
                             const float* __restrict__ W1, const float* __restrict__ b1,
                             const float* __restrict__ g1, const float* __restrict__ be1,
                             const float* __restrict__ W2, const float* __restrict__ b2,
                             const float* __restrict__ g2, const float* __restrict__ be2,
                             const float* __restrict__ W3, const float* __restrict__ b3,
                             float* __restrict__ out) {
  const int row = blockIdx.x;
  const int j   = threadIdx.x;              // 128 threads
  __shared__ float inpS[DD + AAC];
  __shared__ float h1S[H1C];
  __shared__ float h2S[H2C];
  __shared__ float r1[128], r2[128];

  inpS[j]       = z_t[(size_t)row * DD + j];
  inpS[j + 128] = z_t[(size_t)row * DD + 128 + j];
  if (j < AAC) inpS[DD + j] = action[(size_t)row * AAC + j];
  __syncthreads();

  // layer 1: (262 -> 128), LN, relu
  float h = b1[j];
  for (int i = 0; i < DD + AAC; ++i) h += inpS[i] * W1[i * H1C + j];
  r1[j] = h; r2[j] = h * h;
  __syncthreads();
  for (int s = 64; s > 0; s >>= 1) {
    if (j < s) { r1[j] += r1[j + s]; r2[j] += r2[j + s]; }
    __syncthreads();
  }
  {
    const float mu  = r1[0] * (1.0f / 128.0f);
    const float var = r2[0] * (1.0f / 128.0f) - mu * mu;
    const float hn  = (h - mu) * rsqrtf(var + LN_EPS) * g1[j] + be1[j];
    h1S[j] = fmaxf(hn, 0.0f);
  }
  __syncthreads();

  // layer 2: (128 -> 64), LN, relu
  float a2 = 0.0f;
  if (j < H2C) {
    a2 = b2[j];
    for (int i = 0; i < H1C; ++i) a2 += h1S[i] * W2[i * H2C + j];
  }
  r1[j] = (j < H2C) ? a2 : 0.0f;
  r2[j] = (j < H2C) ? a2 * a2 : 0.0f;
  __syncthreads();
  for (int s = 64; s > 0; s >>= 1) {
    if (j < s) { r1[j] += r1[j + s]; r2[j] += r2[j + s]; }
    __syncthreads();
  }
  {
    const float mu  = r1[0] * (1.0f / 64.0f);
    const float var = r2[0] * (1.0f / 64.0f) - mu * mu;
    if (j < H2C) {
      const float hn = (a2 - mu) * rsqrtf(var + LN_EPS) * g2[j] + be2[j];
      h2S[j] = fmaxf(hn, 0.0f);
    }
  }
  __syncthreads();

  // layer 3: (64 -> 256) + squared error
  float sq = 0.0f;
  #pragma unroll
  for (int t = 0; t < 2; ++t) {
    const int c = j + t * 128;
    float zp = b3[c];
    for (int i = 0; i < H2C; ++i) zp += h2S[i] * W3[i * DD + c];
    const float d = zp - z_t1[(size_t)row * DD + c];
    sq += d * d;
  }
  r1[j] = sq;
  __syncthreads();
  for (int s = 64; s > 0; s >>= 1) {
    if (j < s) r1[j] += r1[j + s];
    __syncthreads();
  }
  if (j == 0) out[BQ + row] = r1[0] * (1.0f / (float)DD);
}

// ---------- epistemic (out row 2) ----------
__global__ void epist_kernel(const float* __restrict__ sigma, float* __restrict__ out) {
  const int b = blockIdx.x * blockDim.x + threadIdx.x;
  if (b < BQ) {
    float s = 0.0f;
    #pragma unroll
    for (int i = 0; i < AAC; ++i) s += sigma[(size_t)b * AAC + i];
    out[2 * BQ + b] = s * (1.0f / (float)AAC);
  }
}

#ifdef USE_TDM
// Issue a TDM load of a 128-row x 256-bf16 tile into LDS.
// D# per CDNA5 ISA ch.8: group0 = count/lds_addr/global_addr/type,
// group1 = data_size, tensor dims (dim1 = remaining rows -> HW zero-fills OOB),
// tile dims 256x128, dim0 stride 256. Groups 2/3 zero (2-D tensor).
// This toolchain exposes the 6-arg builtin:
//   (u32x4 g0, i32x8 g1, i32x4 g2, i32x4 g3, i32x8 extra, i32 cpol)
__device__ __forceinline__ void tdm_load_tile(const __bf16* gsrc,
                                              unsigned lds_off, int remRows) {
  const unsigned long long ga = (unsigned long long)(uintptr_t)gsrc;
  u32x4 g0;
  g0.x = 1u;                                               // count=1 (valid user D#)
  g0.y = lds_off;                                          // lds_addr (bytes)
  g0.z = (unsigned)(ga & 0xFFFFFFFFull);                   // global_addr[31:0]
  g0.w = (unsigned)((ga >> 32) & 0x01FFFFFFull) | (2u << 30); // addr[56:32] | type=2

  const unsigned rem = (remRows < 1) ? 1u : (unsigned)remRows;
  i32x8 g1;
  g1[0] = (int)(1u << 16);                                 // data_size = 1 -> 2 bytes
  g1[1] = (int)((256u & 0xFFFFu) << 16);                   // tensor_dim0[15:0] in [63:48]
  g1[2] = (int)(((256u >> 16) & 0xFFFFu) | ((rem & 0xFFFFu) << 16)); // dim0 hi | dim1 lo
  g1[3] = (int)(((rem >> 16) & 0xFFFFu) | (256u << 16));   // dim1 hi | tile_dim0 = 256
  g1[4] = (int)128u;                                       // tile_dim1 = 128, tile_dim2 = 0
  g1[5] = (int)256u;                                       // tensor_dim0_stride[31:0] = 256
  g1[6] = 0;                                               // stride hi | dim1_stride lo
  g1[7] = 0;
  i32x4 z4; z4[0] = 0; z4[1] = 0; z4[2] = 0; z4[3] = 0;
  i32x8 z8; z8[0] = 0; z8[1] = 0; z8[2] = 0; z8[3] = 0;
  z8[4] = 0; z8[5] = 0; z8[6] = 0; z8[7] = 0;
  __builtin_amdgcn_tensor_load_to_lds(g0, g1, z4, z4, z8, 0);
}
#endif

// ---------- main WMMA GEMM + running per-split top-16 ----------
__global__ void __launch_bounds__(256, 1)
sims_topk(const __bf16* __restrict__ zBf, const __bf16* __restrict__ memBf,
          float* __restrict__ partials) {
  // Two 64KB tile buffers (double-buffered TDM). The 128x128 f32 sims tile
  // aliases the *compute* buffer; the TDM always targets the other one.
  __shared__ __align__(16) unsigned char ldsraw[2 * TILE_BYTES];

  const int tid  = threadIdx.x;
  const int wave = tid >> 5;
  const int lane = tid & 31;
  const int hl   = lane >> 4;        // which half-wave
  const int ln15 = lane & 15;
  const int bblk  = blockIdx.x;      // 8 blocks of 128 z-rows
  const int split = blockIdx.y;      // SPLITS splits over M

  // --- A fragments: this wave's 16 z-rows, full K=256, in registers.
  // ISA 16-bit A 16x32 layout: lanes 0-15 hold K{0..7,16..23}, lanes 16-31 hold K{8..15,24..31}.
  const int rowA = bblk * 128 + wave * 16 + ln15;
  const __bf16* zrow = zBf + (size_t)rowA * DD;
  Frag aF[8];
  #pragma unroll
  for (int ks = 0; ks < 8; ++ks) {
    const Q16* pa = (const Q16*)(zrow + ks * 32 + hl * 8);
    aF[ks].q[0] = pa[0];   // elems +0..7
    aF[ks].q[1] = pa[2];   // elems +16..23
  }

  // --- per-thread running top-16 (registers, unrolled cndmask insertion)
  float tk[TOPK];
  float tmin = -1e30f;
  int   tmi  = 0;
  #pragma unroll
  for (int i = 0; i < TOPK; ++i) tk[i] = -1e30f;
  auto insert = [&](float v) {
    if (v > tmin) {
      #pragma unroll
      for (int i = 0; i < TOPK; ++i) if (i == tmi) tk[i] = v;
      tmin = tk[0]; tmi = 0;
      #pragma unroll
      for (int i = 1; i < TOPK; ++i) {
        if (tk[i] < tmin) { tmin = tk[i]; tmi = i; }
      }
    }
  };
  const int trow = tid >> 1;          // 2 threads per z-row for the top-k pass
  const int tcb  = (tid & 1) * 64;

#ifdef USE_TDM
  const unsigned ldsBase = (unsigned)(uintptr_t)(void*)ldsraw; // LDS byte offset
  // prime the pipeline: chunks 0 and 1 in flight
  if (wave == 0) {
    {
      const int mb = (split * CHUNKS_PER_SPLIT + 0) * 128;
      tdm_load_tile(memBf + (size_t)mb * DD, ldsBase, MM - mb);
    }
    if (CHUNKS_PER_SPLIT > 1) {
      const int mb = (split * CHUNKS_PER_SPLIT + 1) * 128;
      tdm_load_tile(memBf + (size_t)mb * DD, ldsBase + TILE_BYTES, MM - mb);
    }
  }
#endif

  for (int c = 0; c < CHUNKS_PER_SPLIT; ++c) {
    const int p = c & 1;
    __bf16* memS  = (__bf16*)(ldsraw + p * TILE_BYTES);
    float*  simsS = (float*)(ldsraw + p * TILE_BYTES);
    const int mbase = (split * CHUNKS_PER_SPLIT + c) * 128;

#ifdef USE_TDM
    if (wave == 0) {
      if (c + 1 < CHUNKS_PER_SPLIT) __builtin_amdgcn_s_wait_tensorcnt(1);
      else                          __builtin_amdgcn_s_wait_tensorcnt(0);
    }
    // warm GL2 for the chunk we will DMA at the end of this iteration
    if (c + 2 < CHUNKS_PER_SPLIT && tid < 128) {
      const int nb = (split * CHUNKS_PER_SPLIT + c + 2) * 128 + tid;
      if (nb < MM) __builtin_prefetch(memBf + (size_t)nb * DD, 0, 1);
    }
#else
    // cooperative stage of the mem tile: 128 rows x 256 bf16 (b128 chunks)
    for (int i = tid; i < 128 * 32; i += 256) {
      const int r = i >> 5, q = i & 31;
      const int gm = mbase + r;
      Q16 val;
      if (gm < MM) val = ((const Q16*)(memBf + (size_t)gm * DD))[q];
      else { val.x = 0u; val.y = 0u; val.z = 0u; val.w = 0u; }
      ((Q16*)(ldsraw + p * TILE_BYTES))[i] = val;
    }
#endif
    __syncthreads();   // tile ready (TDM complete / coop stores done)

    v8f acc[8];
    #pragma unroll
    for (int n = 0; n < 8; ++n)
      acc[n] = (v8f){0.f, 0.f, 0.f, 0.f, 0.f, 0.f, 0.f, 0.f};

    // 8 k-steps x 8 n-tiles of v_wmma_f32_16x16x32_bf16
    #pragma unroll
    for (int ks = 0; ks < 8; ++ks) {
      #pragma unroll
      for (int n = 0; n < 8; ++n) {
        const int brow = n * 16 + ln15;   // B col = mem row; lanes 0-15: K{0..7,16..23}, 16-31: K{8..15,24..31}
        const Q16* pb = (const Q16*)(memS + brow * DD + ks * 32 + hl * 8);
        Frag bF;
        bF.q[0] = pb[0];
        bF.q[1] = pb[2];
        acc[n] = __builtin_amdgcn_wmma_f32_16x16x32_bf16(
            false, aF[ks].v, false, bF.v, (short)0, acc[n], false, false);
      }
    }
    __syncthreads();   // everyone done reading memS -> safe to alias with sims

    // spill sims tile: C layout VGPR j, lanes<16: M=j; lanes>=16: M=j+8; N=lane&15
    #pragma unroll
    for (int n = 0; n < 8; ++n) {
      #pragma unroll
      for (int jj = 0; jj < 8; ++jj) {
        const int r   = wave * 16 + jj + hl * 8;
        const int col = n * 16 + ln15;
        simsS[r * 128 + col] = acc[n][jj];
      }
    }
    __syncthreads();

    // top-k insertion pass (each thread: 64 candidates of its row)
    for (int cc = 0; cc < 64; ++cc) {
      const int col = tcb + cc;
      if (mbase + col < MM) insert(simsS[trow * 128 + col]);
    }
    __syncthreads();   // top-k done -> this buffer may be overwritten

#ifdef USE_TDM
    // refill the buffer we just finished with chunk c+2
    if (wave == 0 && c + 2 < CHUNKS_PER_SPLIT) {
      const int nb = (split * CHUNKS_PER_SPLIT + c + 2) * 128;
      tdm_load_tile(memBf + (size_t)nb * DD, ldsBase + p * TILE_BYTES, MM - nb);
    }
#endif
  }

  // pair-merge the two per-row threads, write per-split partial top-16
  float* mg = (float*)ldsraw;
  if (tid & 1) {
    #pragma unroll
    for (int i = 0; i < TOPK; ++i) mg[trow * TOPK + i] = tk[i];
  }
  __syncthreads();
  if (!(tid & 1)) {
    #pragma unroll
    for (int i = 0; i < TOPK; ++i) insert(mg[trow * TOPK + i]);
    const int gz = bblk * 128 + trow;
    float* dst = partials + ((size_t)gz * SPLITS + split) * TOPK;
    #pragma unroll
    for (int i = 0; i < TOPK; ++i) dst[i] = tk[i];
  }
}

// ---------- merge partial top-16s -> novelty (row 3) and total (row 0) ----------
__global__ void finalize_kernel(const float* __restrict__ partials,
                                float* __restrict__ out) {
  const int row = blockIdx.x;
  const int tid = threadIdx.x;   // 256
  const int NC  = SPLITS * TOPK; // 1936 candidates
  __shared__ float cand[SPLITS * TOPK];
  __shared__ float sv[256];
  __shared__ int   si[256];

  for (int i = tid; i < NC; i += 256)
    cand[i] = partials[(size_t)row * NC + i];
  __syncthreads();

  float sum16 = 0.0f;
  for (int it = 0; it < TOPK; ++it) {
    float bv = -1e30f; int bi = 0;
    for (int i = tid; i < NC; i += 256) {
      const float v = cand[i];
      if (v > bv) { bv = v; bi = i; }
    }
    sv[tid] = bv; si[tid] = bi;
    __syncthreads();
    for (int s = 128; s > 0; s >>= 1) {
      if (tid < s && sv[tid + s] > sv[tid]) { sv[tid] = sv[tid + s]; si[tid] = si[tid + s]; }
      __syncthreads();
    }
    if (tid == 0) { sum16 += sv[0]; cand[si[0]] = -1e30f; }
    __syncthreads();
  }

  if (tid == 0) {
    float nov = 1.0f - sum16 * (1.0f / (float)TOPK);
    nov = fminf(fmaxf(nov, 0.0f), 1.0f);
    out[3 * BQ + row] = nov;
    const float pe = out[BQ + row];
    const float ep = out[2 * BQ + row];
    out[row] = 1.0f * pe + 0.5f * ep + 0.5f * nov;   // W_PRED, W_EPIST, W_NOVEL
  }
}

// ---------- launch ----------
extern "C" void kernel_launch(void* const* d_in, const int* in_sizes, int n_in,
                              void* d_out, int out_size, void* d_ws, size_t ws_size,
                              hipStream_t stream) {
  const float* z_t    = (const float*)d_in[0];
  const float* action = (const float*)d_in[1];
  const float* z_t1   = (const float*)d_in[2];
  const float* sigma  = (const float*)d_in[3];
  const float* mem    = (const float*)d_in[4];
  const float* W1  = (const float*)d_in[5];
  const float* b1  = (const float*)d_in[6];
  const float* g1  = (const float*)d_in[7];
  const float* be1 = (const float*)d_in[8];
  const float* W2  = (const float*)d_in[9];
  const float* b2  = (const float*)d_in[10];
  const float* g2  = (const float*)d_in[11];
  const float* be2 = (const float*)d_in[12];
  const float* W3  = (const float*)d_in[13];
  const float* b3  = (const float*)d_in[14];
  float* out = (float*)d_out;

  // workspace layout: [zBf16 | memBf16 | partials]
  char* ws = (char*)d_ws;
  const size_t zBytes   = (size_t)BQ * DD * 2;          // 512 KB
  const size_t memBytes = (size_t)MM * DD * 2;          // 102.4 MB
  __bf16* zBf    = (__bf16*)ws;
  __bf16* memBf  = (__bf16*)(ws + zBytes);
  float*  parts  = (float*)(ws + zBytes + memBytes);    // 1024*121*16 f32

  prep_norm_bf16<<<BQ, 256, 0, stream>>>(z_t, zBf);
  prep_norm_bf16<<<MM, 256, 0, stream>>>(mem, memBf);
  forward_pred<<<BQ, 128, 0, stream>>>(z_t, action, z_t1,
                                       W1, b1, g1, be1, W2, b2, g2, be2, W3, b3, out);
  epist_kernel<<<(BQ + 255) / 256, 256, 0, stream>>>(sigma, out);
  sims_topk<<<dim3(8, SPLITS), 256, 0, stream>>>(zBf, memBf, parts);
  finalize_kernel<<<BQ, 256, 0, stream>>>(parts, out);
}
